// Head_38955353375556
// MI455X (gfx1250) — matrix operand order repaired and checked by reading, not compiled, not run on hardware
//
#include <hip/hip_runtime.h>

// ---------------------------------------------------------------------------
// Shapes (hardcoded from reference): B=8, T=2048, C=1024, HD=64
// ---------------------------------------------------------------------------
#define BB 8
#define TT 2048
#define CC 1024
#define HD 64
#define NROWS (BB * TT)            // 16384 rows of x
#define SCALE 0.125f               // HD^-0.5 = 1/8

typedef __attribute__((ext_vector_type(16))) __bf16 v16bf;
typedef __attribute__((ext_vector_type(8)))  __bf16 v8bf;
typedef __attribute__((ext_vector_type(8)))  float  v8f;
typedef __attribute__((ext_vector_type(4)))  float  v4f;

__device__ __forceinline__ v16bf join16(v8bf lo, v8bf hi) {
    v16bf r;
#pragma unroll
    for (int i = 0; i < 8; ++i) { r[i] = lo[i]; r[8 + i] = hi[i]; }
    return r;
}

// ---------------------------------------------------------------------------
// Kernel 1: QKV projection.  One wave == one 16-row M tile of [B*T, C].
// A-frag (x rows, f32->bf16) loaded straight from global in WMMA A layout.
// W chunks (32 x 64 per matrix) staged transposed in LDS so B-frags are
// contiguous 32-byte reads.  12 wmma / k-iter, 32 k-iters.
// ---------------------------------------------------------------------------
__global__ void __launch_bounds__(256)
qkv_proj_kernel(const float* __restrict__ x,
                const float* __restrict__ Wq,
                const float* __restrict__ Wk,
                const float* __restrict__ Wv,
                __bf16* __restrict__ Qw,
                __bf16* __restrict__ Kw,
                __bf16* __restrict__ Vw)
{
    // Wt[w][hd_col][k_in_chunk]; row stride 48 elems (96B, 16B-aligned)
    __shared__ __bf16 ldsW[3][64][48];

    const int lane = threadIdx.x & 31;
    const int wave = threadIdx.x >> 5;
    const int tile = blockIdx.x * 8 + wave;   // 16-row tile over B*T
    const int row0 = tile * 16;

    const int arow = row0 + (lane & 15);
    const int aoff = (lane < 16) ? 0 : 16;    // A-layout K sub-offset
    const int ncol = lane & 15;               // B-layout column
    const int kb2  = (lane < 16) ? 0 : 16;    // B-layout K half

    v8f accq[4] = {}; v8f acck[4] = {}; v8f accv[4] = {};

    const float* Ws[3] = { Wq, Wk, Wv };

    for (int kc = 0; kc < CC / 32; ++kc) {
        __syncthreads();
        // ---- cooperative W chunk load + transpose: 256 thr x 8 f32 each ----
        {
            const int t    = threadIdx.x;
            const int krow = t >> 3;              // 0..31 within chunk
            const int h0   = (t & 7) * 8;         // 0..56
#pragma unroll
            for (int w = 0; w < 3; ++w) {
                const v4f* wp = reinterpret_cast<const v4f*>(
                    Ws[w] + (size_t)(kc * 32 + krow) * HD + h0);
                v4f a = wp[0], b = wp[1];
#pragma unroll
                for (int i = 0; i < 4; ++i) {
                    ldsW[w][h0 + i][krow]     = (__bf16)a[i];
                    ldsW[w][h0 + 4 + i][krow] = (__bf16)b[i];
                }
            }
        }
        __syncthreads();

        // ---- A fragment from x (f32 -> bf16), WMMA A layout ----
        v16bf afrag;
        {
            const float* xp = x + (size_t)arow * CC + kc * 32 + aoff;
            const v4f* x4a = reinterpret_cast<const v4f*>(xp);
            const v4f* x4b = reinterpret_cast<const v4f*>(xp + 8);
            v4f a0 = x4a[0], a1 = x4a[1], b0 = x4b[0], b1 = x4b[1];
#pragma unroll
            for (int i = 0; i < 4; ++i) {
                afrag[i]      = (__bf16)a0[i];
                afrag[4 + i]  = (__bf16)a1[i];
                afrag[8 + i]  = (__bf16)b0[i];
                afrag[12 + i] = (__bf16)b1[i];
            }
        }

        // ---- 12 WMMAs: 4 n-tiles x {Q,K,V} ----
#pragma unroll
        for (int nt = 0; nt < 4; ++nt) {
            const __bf16* bq = &ldsW[0][nt * 16 + ncol][kb2];
            const __bf16* bk = &ldsW[1][nt * 16 + ncol][kb2];
            const __bf16* bv = &ldsW[2][nt * 16 + ncol][kb2];
            const v8bf* q8 = reinterpret_cast<const v8bf*>(bq);
            const v8bf* k8 = reinterpret_cast<const v8bf*>(bk);
            const v8bf* v8 = reinterpret_cast<const v8bf*>(bv);
            v16bf bfq = join16(q8[0], q8[1]);
            v16bf bfk = join16(k8[0], k8[1]);
            v16bf bfv = join16(v8[0], v8[1]);
            accq[nt] = __builtin_amdgcn_wmma_f32_16x16x32_bf16(
                false, afrag, false, bfq, (short)0, accq[nt], false, false);
            acck[nt] = __builtin_amdgcn_wmma_f32_16x16x32_bf16(
                false, afrag, false, bfk, (short)0, acck[nt], false, false);
            accv[nt] = __builtin_amdgcn_wmma_f32_16x16x32_bf16(
                false, afrag, false, bfv, (short)0, accv[nt], false, false);
        }
    }

    // ---- store D (C/D layout: vgpr r, lane half -> row; lane&15 -> col) ----
    const int mh = (lane < 16) ? 0 : 8;
#pragma unroll
    for (int nt = 0; nt < 4; ++nt) {
#pragma unroll
        for (int r = 0; r < 8; ++r) {
            const size_t idx = (size_t)(row0 + mh + r) * HD + nt * 16 + (lane & 15);
            Qw[idx] = (__bf16)accq[nt][r];
            Kw[idx] = (__bf16)acck[nt][r];
            Vw[idx] = (__bf16)accv[nt][r];
        }
    }
}

// ---------------------------------------------------------------------------
// Kernel 2: causal flash attention.  One wave == 16 query rows; 8 independent
// waves / block (no inter-wave barriers; per-wave LDS slices).
// Key blocks of 32: S = Q K^T (K^T B-frags contiguous from global row-major
// K), online softmax with shfl row reductions, P staged through LDS to get
// the A layout, V staged transposed in LDS so PV B-frags are contiguous.
// ---------------------------------------------------------------------------
#define VSTR 40        // Vt row stride (elems); 80B, 16B aligned
#define PSTR 40        // P row stride

__global__ void __launch_bounds__(256)
attn_kernel(const __bf16* __restrict__ Qw,
            const __bf16* __restrict__ Kw,
            const __bf16* __restrict__ Vw,
            float* __restrict__ out)
{
    __shared__ __bf16 lds[8][HD * VSTR + 16 * PSTR];   // 6400 B per wave

    const int lane = threadIdx.x & 31;
    const int wave = threadIdx.x >> 5;
    const int tile = blockIdx.x * 8 + wave;     // over B*T/16 = 1024 tiles
    const int b    = tile >> 7;                 // 128 tiles per batch
    const int q0   = (tile & 127) * 16;         // local query row0 in batch

    __bf16* vt   = lds[wave];                   // Vt[h][k]  (64 x 32)
    __bf16* pbuf = lds[wave] + HD * VSTR;       // P[m][n]   (16 x 32)

    const __bf16* Qb = Qw + (size_t)tile * 16 * HD;
    const __bf16* Kb = Kw + (size_t)b * TT * HD;
    const __bf16* Vb = Vw + (size_t)b * TT * HD;

    const int mh   = (lane < 16) ? 0 : 8;       // C/D row half
    const int aoff = (lane < 16) ? 0 : 16;      // A-layout K sub-offset
    const int kb2  = (lane < 16) ? 0 : 16;      // B-layout K half
    const int ncol = lane & 15;

    // ---- Q fragments (2 K-chunks over HD=64), straight from global ----
    v16bf qf[2];
#pragma unroll
    for (int c = 0; c < 2; ++c) {
        const __bf16* qp = Qb + (size_t)(lane & 15) * HD + c * 32 + aoff;
        const v8bf* qa = reinterpret_cast<const v8bf*>(qp);
        const v8bf* qb2 = reinterpret_cast<const v8bf*>(qp + 8);
        qf[c] = join16(qa[0], qb2[0]);
    }

    const float NEG_INF = -__builtin_inff();
    float mrow[8], lrow[8];
#pragma unroll
    for (int r = 0; r < 8; ++r) { mrow[r] = NEG_INF; lrow[r] = 0.0f; }
    v8f oacc[4] = {};

    const int nkb = ((q0 + 15) >> 5) + 1;       // causal key-block count

    for (int kb = 0; kb < nkb; ++kb) {
        const int k0 = kb * 32;

        // ---- stage V tile transposed: lane owns key row (k0+lane) ----
        {
            const v8bf* vp = reinterpret_cast<const v8bf*>(
                Vb + (size_t)(k0 + lane) * HD);
#pragma unroll
            for (int c8 = 0; c8 < 8; ++c8) {
                v8bf vv = vp[c8];
#pragma unroll
                for (int i = 0; i < 8; ++i)
                    vt[(c8 * 8 + i) * VSTR + lane] = vv[i];
            }
        }

        // ---- S = Q K^T : 2 n-tiles x 2 K-chunks ----
        v8f s[2] = {};
#pragma unroll
        for (int nt = 0; nt < 2; ++nt) {
#pragma unroll
            for (int c = 0; c < 2; ++c) {
                const __bf16* kp = Kb + (size_t)(k0 + nt * 16 + ncol) * HD
                                      + c * 32 + kb2;
                const v8bf* k8 = reinterpret_cast<const v8bf*>(kp);
                v16bf bfrag = join16(k8[0], k8[1]);
                s[nt] = __builtin_amdgcn_wmma_f32_16x16x32_bf16(
                    false, qf[c], false, bfrag, (short)0, s[nt], false, false);
            }
            s[nt] *= SCALE;
        }

        // ---- causal mask (element (m,n): m = mh+r, n = lane&15) ----
#pragma unroll
        for (int nt = 0; nt < 2; ++nt) {
            const int j = k0 + nt * 16 + ncol;
#pragma unroll
            for (int r = 0; r < 8; ++r) {
                if (j > q0 + mh + r) s[nt][r] = NEG_INF;
            }
        }

        // ---- online softmax: row max / exp / row sum (shfl in 16-lane half)
        float alpha[8];
#pragma unroll
        for (int r = 0; r < 8; ++r) {
            float mx = fmaxf(s[0][r], s[1][r]);
#pragma unroll
            for (int d = 1; d < 16; d <<= 1)
                mx = fmaxf(mx, __shfl_xor(mx, d, 32));
            const float mnew = fmaxf(mrow[r], mx);
            alpha[r] = __expf(mrow[r] - mnew);
            mrow[r] = mnew;
            const float p0 = __expf(s[0][r] - mnew);
            const float p1 = __expf(s[1][r] - mnew);
            s[0][r] = p0; s[1][r] = p1;
            float rs = p0 + p1;
#pragma unroll
            for (int d = 1; d < 16; d <<= 1)
                rs += __shfl_xor(rs, d, 32);
            lrow[r] = lrow[r] * alpha[r] + rs;
        }

        // ---- rescale output accumulators ----
#pragma unroll
        for (int nt = 0; nt < 4; ++nt)
#pragma unroll
            for (int r = 0; r < 8; ++r) oacc[nt][r] *= alpha[r];

        // ---- P: C-layout -> LDS -> A-layout fragment ----
#pragma unroll
        for (int nt = 0; nt < 2; ++nt) {
            const int n = nt * 16 + ncol;
#pragma unroll
            for (int r = 0; r < 8; ++r)
                pbuf[(mh + r) * PSTR + n] = (__bf16)s[nt][r];
        }
        v16bf pf;
        {
            const __bf16* pp = pbuf + (lane & 15) * PSTR + aoff;
            const v8bf* p8a = reinterpret_cast<const v8bf*>(pp);
            const v8bf* p8b = reinterpret_cast<const v8bf*>(pp + 8);
            pf = join16(p8a[0], p8b[0]);
        }

        // ---- O += P V : 4 hd n-tiles, K=32 in one wmma ----
#pragma unroll
        for (int nt = 0; nt < 4; ++nt) {
            const __bf16* vp2 = vt + (size_t)(nt * 16 + ncol) * VSTR + kb2;
            const v8bf* v8 = reinterpret_cast<const v8bf*>(vp2);
            v16bf bfrag = join16(v8[0], v8[1]);
            oacc[nt] = __builtin_amdgcn_wmma_f32_16x16x32_bf16(
                false, pf, false, bfrag, (short)0, oacc[nt], false, false);
        }
    }

    // ---- epilogue: O / l, fp32 store ----
#pragma unroll
    for (int nt = 0; nt < 4; ++nt) {
#pragma unroll
        for (int r = 0; r < 8; ++r) {
            const size_t row = (size_t)tile * 16 + mh + r;
            out[row * HD + nt * 16 + ncol] = oacc[nt][r] / lrow[r];
        }
    }
}

// ---------------------------------------------------------------------------
extern "C" void kernel_launch(void* const* d_in, const int* in_sizes, int n_in,
                              void* d_out, int out_size, void* d_ws, size_t ws_size,
                              hipStream_t stream) {
    const float* x  = (const float*)d_in[0];
    const float* Wq = (const float*)d_in[1];
    const float* Wk = (const float*)d_in[2];
    const float* Wv = (const float*)d_in[3];

    __bf16* Qw = (__bf16*)d_ws;                       // 2 MB
    __bf16* Kw = Qw + (size_t)NROWS * HD;             // 2 MB
    __bf16* Vw = Kw + (size_t)NROWS * HD;             // 2 MB

    // 1024 16-row tiles, 8 waves (256 threads) per block
    qkv_proj_kernel<<<NROWS / 16 / 8, 256, 0, stream>>>(x, Wq, Wk, Wv, Qw, Kw, Vw);
    attn_kernel<<<NROWS / 16 / 8, 256, 0, stream>>>(Qw, Kw, Vw, (float*)d_out);
}
